// PointNetPlusPlusAttnFusion_cross_73194832658940
// MI455X (gfx1250) — compile-verified, matrix operand order checked by reading, not run on hardware
//
#include <hip/hip_runtime.h>
#include <hip/hip_bf16.h>
#include <math.h>

// ---------------------------------------------------------------------------
// PointNet++ + cross-attention forward for MI455X (gfx1250, wave32, WMMA).
// Dense layers: f16-input / f32-accumulate WMMA GEMM (v_wmma_f32_16x16x32_f16),
// block tile 128x64, 32x32 per wave (4 WMMAs / 8 fragment loads per K-step),
// branch-free staging, packed ds_store_b128, LDS double buffering with one
// barrier per K step.
// ---------------------------------------------------------------------------

typedef __attribute__((ext_vector_type(16))) _Float16 v16h_t;
typedef __attribute__((ext_vector_type(8)))  _Float16 v8h_t;
typedef __attribute__((ext_vector_type(8)))  float    v8f_t;

#define TILE_K     32
#define LDS_STRIDE 40   // 32 halves + 8 pad -> 80B row stride, 16B aligned

// ------------------------- WMMA GEMM: C = A*W (+bias) ----------------------
// A: [M,K] row-major f32, W: [K,N] row-major f32, C: [M,N] f32.
// REQUIREMENT: M % 128 == 0 and N % 64 == 0 (true for every call site here);
// K is arbitrary (zero-padded per 32-wide slab).
// Block tile 128x64, 256 threads = 8 waves; wave w owns the 32x32 quadrant at
// rows 32*(w&3), cols 32*(w>>2): fragments a0,a1 x b0,b1 -> 4 accumulators.
__global__ __launch_bounds__(256)
void gemm_bias_kernel(const float* __restrict__ A, const float* __restrict__ W,
                      const float* __restrict__ bias, float* __restrict__ C,
                      int M, int N, int K)
{
    __shared__ __align__(16) _Float16 As[2][128 * LDS_STRIDE];
    __shared__ __align__(16) _Float16 Bs[2][64 * LDS_STRIDE];   // Bs[n][k] = W[k][n]

    const int tid    = threadIdx.x;
    const int wave   = tid >> 5;
    const int lane   = tid & 31;
    const int blockM = blockIdx.x * 128;
    const int blockN = blockIdx.y * 64;

    // ---- staging maps (branch-free, packed 8-half chunks) ----
    const int rowA = tid >> 2;            // 0..63 (and rowA+64)
    const int colA = (tid & 3) * 8;       // 0,8,16,24
    const int nB   = tid & 63;            // 0..63   (W tile column)
    const int kB   = (tid >> 6) * 8;      // 0,8,16,24

    const float* aRow0 = A + (size_t)(blockM + rowA) * K;
    const float* aRow1 = A + (size_t)(blockM + rowA + 64) * K;
    const float* wCol  = W + blockN + nB;

    // ---- per-wave fragment addresses (constant across K loop) ----
    const int tm   = (wave & 3) * 32;
    const int tn   = (wave >> 2) * 32;
    const int lm   = lane & 15;
    const int koff = (lane < 16) ? 0 : 8;
    const int a0Off = (tm + lm)      * LDS_STRIDE + koff;
    const int a1Off = (tm + 16 + lm) * LDS_STRIDE + koff;
    const int b0Off = (tn + lm)      * LDS_STRIDE + koff;
    const int b1Off = (tn + 16 + lm) * LDS_STRIDE + koff;

    float ar0[8], ar1[8], br[8];

    auto stage_load = [&](int k0) {
        #pragma unroll
        for (int i = 0; i < 8; ++i) {               // A slabs: contiguous in K
            int gk  = k0 + colA + i;
            int gkc = gk < K ? gk : K - 1;          // clamp, no exec branch
            float v0 = aRow0[gkc];
            float v1 = aRow1[gkc];
            ar0[i] = gk < K ? v0 : 0.0f;
            ar1[i] = gk < K ? v1 : 0.0f;
        }
        #pragma unroll
        for (int i = 0; i < 8; ++i) {               // W slab: strided by N, coalesced across lanes
            int gk  = k0 + kB + i;
            int gkc = gk < K ? gk : K - 1;
            float v = wCol[(size_t)gkc * N];
            br[i] = gk < K ? v : 0.0f;
        }
        if (k0 + TILE_K < K) {                      // prefetch tile-after-next
            __builtin_prefetch(&aRow0[k0 + TILE_K + colA], 0, 1);
            __builtin_prefetch(&aRow1[k0 + TILE_K + colA], 0, 1);
            __builtin_prefetch(&wCol[(size_t)(k0 + TILE_K + kB) * N], 0, 1);
        }
    };
    auto stage_store = [&](int buf) {
        v8h_t a0h, a1h, bh;
        #pragma unroll
        for (int i = 0; i < 8; ++i) {
            a0h[i] = (_Float16)ar0[i];
            a1h[i] = (_Float16)ar1[i];
            bh[i]  = (_Float16)br[i];
        }
        *(v8h_t*)&As[buf][(rowA)      * LDS_STRIDE + colA] = a0h;  // ds_store_b128
        *(v8h_t*)&As[buf][(rowA + 64) * LDS_STRIDE + colA] = a1h;  // ds_store_b128
        *(v8h_t*)&Bs[buf][nB          * LDS_STRIDE + kB  ] = bh;   // ds_store_b128
    };

    v8f_t acc00 = {}, acc01 = {}, acc10 = {}, acc11 = {};
    const int ntiles = (K + TILE_K - 1) / TILE_K;

    stage_load(0);
    stage_store(0);
    __syncthreads();

    for (int t = 0; t < ntiles; ++t) {
        const int cur = t & 1;
        const int nxt = cur ^ 1;
        if (t + 1 < ntiles) stage_load((t + 1) * TILE_K);

        // ---- fragments from current buffer (ISA 16-bit A/B layout) ----
        const _Float16* pa0 = &As[cur][a0Off];
        const _Float16* pa1 = &As[cur][a1Off];
        const _Float16* pb0 = &Bs[cur][b0Off];
        const _Float16* pb1 = &Bs[cur][b1Off];
        v8h_t a0l = *(const v8h_t*)(pa0), a0h = *(const v8h_t*)(pa0 + 16);
        v8h_t a1l = *(const v8h_t*)(pa1), a1h = *(const v8h_t*)(pa1 + 16);
        v8h_t b0l = *(const v8h_t*)(pb0), b0h = *(const v8h_t*)(pb0 + 16);
        v8h_t b1l = *(const v8h_t*)(pb1), b1h = *(const v8h_t*)(pb1 + 16);

        v16h_t af0, af1, bf0, bf1;
        #pragma unroll
        for (int i = 0; i < 8; ++i) {
            af0[i] = a0l[i]; af0[8 + i] = a0h[i];
            af1[i] = a1l[i]; af1[8 + i] = a1h[i];
            bf0[i] = b0l[i]; bf0[8 + i] = b0h[i];
            bf1[i] = b1l[i]; bf1[8 + i] = b1h[i];
        }

        // 4 independent accumulation chains -> good WMMA pipe utilization
        acc00 = __builtin_amdgcn_wmma_f32_16x16x32_f16(false, af0, false, bf0,
                                                       (short)0, acc00, false, false);
        acc01 = __builtin_amdgcn_wmma_f32_16x16x32_f16(false, af0, false, bf1,
                                                       (short)0, acc01, false, false);
        acc10 = __builtin_amdgcn_wmma_f32_16x16x32_f16(false, af1, false, bf0,
                                                       (short)0, acc10, false, false);
        acc11 = __builtin_amdgcn_wmma_f32_16x16x32_f16(false, af1, false, bf1,
                                                       (short)0, acc11, false, false);

        if (t + 1 < ntiles) stage_store(nxt);
        __syncthreads();
    }

    // ---- epilogue: lane n = lane&15, rows r + 8*(lane>=16); M%128==0, N%64==0 ----
    const int mrow0 = blockM + tm + (lane >> 4) * 8;
    const int mrow1 = mrow0 + 16;
    const int n0    = blockN + tn + lm;
    const int n1    = n0 + 16;
    const float b0v = bias ? bias[n0] : 0.0f;
    const float b1v = bias ? bias[n1] : 0.0f;
    #pragma unroll
    for (int r = 0; r < 8; ++r) {
        const size_t m0 = (size_t)(mrow0 + r);
        const size_t m1 = (size_t)(mrow1 + r);
        C[m0 * N + n0] = acc00[r] + b0v;
        C[m0 * N + n1] = acc01[r] + b1v;
        C[m1 * N + n0] = acc10[r] + b0v;
        C[m1 * N + n1] = acc11[r] + b1v;
    }
}

// ---------------------- farthest point sampling -----------------------------
// One block per batch. Distance array lives in LDS; 'npoint' serial argmax
// rounds (this is the latency wall of the whole net, not the GEMMs).
__global__ void fps_kernel(const float* __restrict__ xyz, int* __restrict__ out_idx,
                           int Nn, int npoint)
{
    const int b = blockIdx.x;
    const float* p = xyz + (size_t)b * Nn * 3;
    __shared__ float dist[4096];
    __shared__ float red_v[256];
    __shared__ int   red_i[256];
    __shared__ int   s_cur;

    for (int i = threadIdx.x; i < Nn; i += blockDim.x) dist[i] = 1e10f;
    if (threadIdx.x == 0) s_cur = 0;
    __syncthreads();

    for (int it = 0; it < npoint; ++it) {
        const int c = s_cur;
        if (threadIdx.x == 0) out_idx[b * npoint + it] = c;
        const float cx = p[c * 3 + 0], cy = p[c * 3 + 1], cz = p[c * 3 + 2];
        float best = -1.0f; int bi = 0;
        for (int i = threadIdx.x; i < Nn; i += blockDim.x) {
            float dx = p[i * 3 + 0] - cx;
            float dy = p[i * 3 + 1] - cy;
            float dz = p[i * 3 + 2] - cz;
            float d  = dx * dx + dy * dy + dz * dz;
            float nd = fminf(dist[i], d);
            dist[i] = nd;
            if (nd > best) { best = nd; bi = i; }
        }
        red_v[threadIdx.x] = best;
        red_i[threadIdx.x] = bi;
        __syncthreads();
        for (int s = 128; s > 0; s >>= 1) {
            if (threadIdx.x < s) {
                float v2 = red_v[threadIdx.x + s];
                int   i2 = red_i[threadIdx.x + s];
                if (v2 > red_v[threadIdx.x] ||
                    (v2 == red_v[threadIdx.x] && i2 < red_i[threadIdx.x])) {
                    red_v[threadIdx.x] = v2;
                    red_i[threadIdx.x] = i2;
                }
            }
            __syncthreads();
        }
        if (threadIdx.x == 0) s_cur = red_i[0];
        __syncthreads();
    }
}

// gather xyz rows by index: out[b,s,:] = src[b, idx[b,s], :]
__global__ void gather3_kernel(const float* __restrict__ src, const int* __restrict__ idx,
                               float* __restrict__ out, int Nn, int S, int total)
{
    int t = blockIdx.x * blockDim.x + threadIdx.x;
    if (t >= total) return;
    int b = t / S;
    int j = idx[t];
    const float* pi = src + ((size_t)b * Nn + j) * 3;
    float* po = out + (size_t)t * 3;
    po[0] = pi[0]; po[1] = pi[1]; po[2] = pi[2];
}

// ball query: first 'nsample' indices (ascending) within radius, pad with first
__global__ void query_ball_kernel(const float* __restrict__ xyz,
                                  const float* __restrict__ new_xyz,
                                  int* __restrict__ idx,
                                  int Nn, int S, int nsample, float r2, int total)
{
    int t = blockIdx.x * blockDim.x + threadIdx.x;
    if (t >= total) return;                    // total = B*S
    int b = t / S;
    const float* nx = new_xyz + (size_t)t * 3;
    float cx = nx[0], cy = nx[1], cz = nx[2];
    int* o = idx + (size_t)t * nsample;
    int cnt = 0, first = -1;
    for (int j = 0; j < Nn && cnt < nsample; ++j) {
        const float* pj = xyz + ((size_t)b * Nn + j) * 3;
        float dx = pj[0] - cx, dy = pj[1] - cy, dz = pj[2] - cz;
        if (dx * dx + dy * dy + dz * dz <= r2) {
            if (first < 0) first = j;
            o[cnt++] = j;
        }
    }
    if (first < 0) first = 0;
    for (; cnt < nsample; ++cnt) o[cnt] = first;
}

// grouped = concat(xyz[idx]-new_xyz, points[idx]) -> rows (B*S*ns) x (3+Cin)
__global__ void group_concat_kernel(const float* __restrict__ xyz,
                                    const float* __restrict__ new_xyz,
                                    const float* __restrict__ pts,
                                    const int* __restrict__ idx,
                                    float* __restrict__ out,
                                    int Nn, int S, int ns, int Cin, int total)
{
    int t = blockIdx.x * blockDim.x + threadIdx.x;
    if (t >= total) return;                    // total = B*S*ns
    int bs = t / ns;
    int b  = bs / S;
    int j  = idx[t];
    const float* pj = xyz + ((size_t)b * Nn + j) * 3;
    const float* nc = new_xyz + (size_t)bs * 3;
    float* o = out + (size_t)t * (3 + Cin);
    o[0] = pj[0] - nc[0];
    o[1] = pj[1] - nc[1];
    o[2] = pj[2] - nc[2];
    const float* fj = pts + ((size_t)b * Nn + j) * Cin;
    for (int c = 0; c < Cin; ++c) o[3 + c] = fj[c];
}

// per-channel mean/var over all rows (matches jnp mean/var over batch axes)
__global__ void col_stats_kernel(const float* __restrict__ X, float* __restrict__ mean,
                                 float* __restrict__ var, int M, int N)
{
    int n = blockIdx.x;
    float s = 0.f, s2 = 0.f;
    for (int i = threadIdx.x; i < M; i += blockDim.x) {
        float v = X[(size_t)i * N + n];
        s += v; s2 += v * v;
    }
    __shared__ float rs[256], rq[256];
    rs[threadIdx.x] = s; rq[threadIdx.x] = s2;
    __syncthreads();
    for (int t = 128; t > 0; t >>= 1) {
        if (threadIdx.x < t) {
            rs[threadIdx.x] += rs[threadIdx.x + t];
            rq[threadIdx.x] += rq[threadIdx.x + t];
        }
        __syncthreads();
    }
    if (threadIdx.x == 0) {
        float m = rs[0] / (float)M;
        mean[n] = m;
        var[n]  = fmaxf(rq[0] / (float)M - m * m, 0.0f);
    }
}

__global__ void bn_relu_kernel(float* __restrict__ X, const float* __restrict__ mean,
                               const float* __restrict__ var, const float* __restrict__ g,
                               const float* __restrict__ beta, int N, size_t total)
{
    size_t t = (size_t)blockIdx.x * blockDim.x + threadIdx.x;
    if (t >= total) return;
    int n = (int)(t % N);
    float x = X[t];
    x = (x - mean[n]) * rsqrtf(var[n] + 1e-5f) * g[n] + beta[n];
    X[t] = fmaxf(x, 0.0f);
}

// max over the sample axis: in (B*S*ns, C) -> out (B*S, C)
__global__ void maxpool_kernel(const float* __restrict__ X, float* __restrict__ Y,
                               int ns, int C, int totalOut)
{
    int t = blockIdx.x * blockDim.x + threadIdx.x;
    if (t >= totalOut) return;
    int c  = t % C;
    int rs = t / C;
    const float* p = X + ((size_t)rs * ns) * C + c;
    float m = p[0];
    for (int k = 1; k < ns; ++k) m = fmaxf(m, p[(size_t)k * C]);
    Y[t] = m;
}

// ----------------------- cross attention pieces -----------------------------
__global__ void attn_scores_kernel(const float* __restrict__ q, const float* __restrict__ k,
                                   float* __restrict__ sc, int S, int T, int C,
                                   float scale, int total)
{
    int id = blockIdx.x * blockDim.x + threadIdx.x;
    if (id >= total) return;                  // total = B*S*T
    int t  = id % T;
    int bs = id / T;
    int b  = bs / S;
    const float* qr = q + (size_t)bs * C;
    const float* kr = k + ((size_t)b * T + t) * C;
    float acc = 0.f;
    for (int c = 0; c < C; ++c) acc += qr[c] * kr[c];
    sc[id] = acc * scale;
}

__global__ void softmax_rows_kernel(float* __restrict__ sc, int T)
{
    float* row = sc + (size_t)blockIdx.x * T;
    int t = threadIdx.x;                      // blockDim.x == T == 128
    __shared__ float red[128];
    float v = row[t];
    red[t] = v; __syncthreads();
    for (int s = 64; s > 0; s >>= 1) {
        if (t < s) red[t] = fmaxf(red[t], red[t + s]);
        __syncthreads();
    }
    float m = red[0]; __syncthreads();
    float e = expf(v - m);
    red[t] = e; __syncthreads();
    for (int s = 64; s > 0; s >>= 1) {
        if (t < s) red[t] += red[t + s];
        __syncthreads();
    }
    row[t] = e / red[0];
}

__global__ void attn_apply_kernel(const float* __restrict__ sc, const float* __restrict__ v,
                                  float* __restrict__ out, int S, int T, int C, int total)
{
    int id = blockIdx.x * blockDim.x + threadIdx.x;
    if (id >= total) return;                  // total = B*S*C
    int c  = id % C;
    int bs = id / C;
    int b  = bs / S;
    const float* sr = sc + (size_t)bs * T;
    const float* vb = v + (size_t)b * T * C + c;
    float acc = 0.f;
    for (int t = 0; t < T; ++t) acc += sr[t] * vb[(size_t)t * C];
    out[id] = acc;
}

// -------------------- 3-NN inverse-distance interpolation -------------------
__global__ void three_interp_kernel(const float* __restrict__ xyz1,
                                    const float* __restrict__ xyz2,
                                    const float* __restrict__ pts2,
                                    float* __restrict__ out,
                                    int N1, int N2, int C, int total)
{
    int t = blockIdx.x * blockDim.x + threadIdx.x;
    if (t >= total) return;                   // total = B*N1
    int b = t / N1;
    const float* p = xyz1 + (size_t)t * 3;
    float px = p[0], py = p[1], pz = p[2];
    float d0 = 1e30f, d1 = 1e30f, d2 = 1e30f;
    int   i0 = 0, i1 = 0, i2 = 0;
    const float* q = xyz2 + (size_t)b * N2 * 3;
    for (int j = 0; j < N2; ++j) {
        float dx = q[j * 3 + 0] - px;
        float dy = q[j * 3 + 1] - py;
        float dz = q[j * 3 + 2] - pz;
        float d = dx * dx + dy * dy + dz * dz;
        if (d < d0)      { d2 = d1; i2 = i1; d1 = d0; i1 = i0; d0 = d; i0 = j; }
        else if (d < d1) { d2 = d1; i2 = i1; d1 = d;  i1 = j; }
        else if (d < d2) { d2 = d;  i2 = j; }
    }
    float w0 = 1.f / (d0 + 1e-8f), w1 = 1.f / (d1 + 1e-8f), w2 = 1.f / (d2 + 1e-8f);
    float ws = w0 + w1 + w2;
    w0 /= ws; w1 /= ws; w2 /= ws;
    const float* r0 = pts2 + ((size_t)b * N2 + i0) * C;
    const float* r1 = pts2 + ((size_t)b * N2 + i1) * C;
    const float* r2 = pts2 + ((size_t)b * N2 + i2) * C;
    float* o = out + (size_t)t * C;
    for (int c = 0; c < C; ++c) o[c] = w0 * r0[c] + w1 * r1[c] + w2 * r2[c];
}

// out[row] = concat(A[row] (C1), B[row] (C2))
__global__ void concat_rows_kernel(const float* __restrict__ A, const float* __restrict__ Bb,
                                   float* __restrict__ out, int C1, int C2, int rows)
{
    int t = blockIdx.x * blockDim.x + threadIdx.x;
    if (t >= rows) return;
    const float* a = A  + (size_t)t * C1;
    const float* b = Bb + (size_t)t * C2;
    float* o = out + (size_t)t * (C1 + C2);
    for (int c = 0; c < C1; ++c) o[c]      = a[c];
    for (int c = 0; c < C2; ++c) o[C1 + c] = b[c];
}

// ----------------------------- host orchestration ---------------------------
struct Layer { const float *W, *b, *g, *beta; };

static inline void run_mlp_layer(const float* X, const Layer& L, float* Y, float* stats,
                                 int M, int K, int N, hipStream_t s)
{
    dim3 grid((M + 127) / 128, (N + 63) / 64);
    gemm_bias_kernel<<<grid, 256, 0, s>>>(X, L.W, L.b, Y, M, N, K);
    col_stats_kernel<<<N, 256, 0, s>>>(Y, stats, stats + N, M, N);
    size_t total = (size_t)M * N;
    bn_relu_kernel<<<(unsigned)((total + 255) / 256), 256, 0, s>>>(Y, stats, stats + N,
                                                                   L.g, L.beta, N, total);
}

extern "C" void kernel_launch(void* const* d_in, const int* in_sizes, int n_in,
                              void* d_out, int out_size, void* d_ws, size_t ws_size,
                              hipStream_t stream)
{
    (void)in_sizes; (void)n_in; (void)out_size; (void)ws_size;
    const int Nn = 4096;

    // Input order: xyz, xyz2, then params flattened with jax dict-key sort:
    // attn(Wq,Wk,Wv,Wo), fp1(3 layers x (W,b,g,beta)), fp2(2x4), sa1(3x4), sa2(3x4)
    const float* xyzIn[2] = { (const float*)d_in[0], (const float*)d_in[1] };
    const float* Wq = (const float*)d_in[2];
    const float* Wk = (const float*)d_in[3];
    const float* Wv = (const float*)d_in[4];
    const float* Wo = (const float*)d_in[5];
    auto layer = [&](int base) -> Layer {
        return Layer{ (const float*)d_in[base],     (const float*)d_in[base + 1],
                      (const float*)d_in[base + 2], (const float*)d_in[base + 3] };
    };
    Layer fp1l[3] = { layer(6),  layer(10), layer(14) };
    Layer fp2l[2] = { layer(18), layer(22) };
    Layer sa1l[3] = { layer(26), layer(30), layer(34) };
    Layer sa2l[3] = { layer(38), layer(42), layer(46) };

    // bump allocator over d_ws
    char* base = (char*)d_ws;
    size_t off = 0;
    auto allocf = [&](size_t n) -> float* {
        float* p = (float*)(base + off);
        off += n * sizeof(float);
        off = (off + 255) & ~(size_t)255;
        return p;
    };
    auto alloci = [&](size_t n) -> int* {
        int* p = (int*)(base + off);
        off += n * sizeof(int);
        off = (off + 255) & ~(size_t)255;
        return p;
    };

    float *l1x[2], *l1p[2], *l2x[2], *l2p[2], *l2c[2];
    for (int s2 = 0; s2 < 2; ++s2) {
        l1x[s2] = allocf((size_t)16 * 512 * 3);
        l1p[s2] = allocf((size_t)16 * 512 * 128);
        l2x[s2] = allocf((size_t)16 * 128 * 3);
        l2p[s2] = allocf((size_t)16 * 128 * 256);
        l2c[s2] = allocf((size_t)16 * 128 * 512);
    }
    float* stats  = allocf(1024);
    int*   fpsIdx = alloci((size_t)16 * 512);
    int*   ballIdx = alloci((size_t)16 * 512 * 32);
    float* qb  = allocf((size_t)2048 * 256);
    float* kb  = allocf((size_t)2048 * 256);
    float* vb  = allocf((size_t)2048 * 256);
    float* scb = allocf((size_t)16 * 128 * 128);
    float* avb = allocf((size_t)2048 * 256);
    float* aob = allocf((size_t)2048 * 256);
    float* l1b = allocf((size_t)16 * 512 * 256);
    float* bufA = allocf((size_t)33554432);     // 128 MB ping
    float* bufB = allocf((size_t)33554432);     // 128 MB pong

    // ------------------------------ encoders --------------------------------
    for (int s2 = 0; s2 < 2; ++s2) {
        const float* xyz = xyzIn[s2];
        // SA1: npoint=512, r=0.2, ns=32, chans 6->64->64->128
        fps_kernel<<<16, 256, 0, stream>>>(xyz, fpsIdx, Nn, 512);
        int totS = 16 * 512;
        gather3_kernel<<<(totS + 255) / 256, 256, 0, stream>>>(xyz, fpsIdx, l1x[s2], Nn, 512, totS);
        query_ball_kernel<<<(totS + 255) / 256, 256, 0, stream>>>(xyz, l1x[s2], ballIdx,
                                                                  Nn, 512, 32, 0.04f, totS);
        int totG = 16 * 512 * 32;
        group_concat_kernel<<<(totG + 255) / 256, 256, 0, stream>>>(xyz, l1x[s2], xyz, ballIdx,
                                                                    bufA, Nn, 512, 32, 3, totG);
        run_mlp_layer(bufA, sa1l[0], bufB, stats, totG, 6, 64, stream);
        run_mlp_layer(bufB, sa1l[1], bufA, stats, totG, 64, 64, stream);
        run_mlp_layer(bufA, sa1l[2], bufB, stats, totG, 64, 128, stream);
        int totP1 = 16 * 512 * 128;
        maxpool_kernel<<<(totP1 + 255) / 256, 256, 0, stream>>>(bufB, l1p[s2], 32, 128, totP1);

        // SA2: npoint=128, r=0.4, ns=64, chans 131->128->128->256
        fps_kernel<<<16, 256, 0, stream>>>(l1x[s2], fpsIdx, 512, 128);
        int totS2 = 16 * 128;
        gather3_kernel<<<(totS2 + 255) / 256, 256, 0, stream>>>(l1x[s2], fpsIdx, l2x[s2], 512, 128, totS2);
        query_ball_kernel<<<(totS2 + 255) / 256, 256, 0, stream>>>(l1x[s2], l2x[s2], ballIdx,
                                                                   512, 128, 64, 0.16f, totS2);
        int totG2 = 16 * 128 * 64;
        group_concat_kernel<<<(totG2 + 255) / 256, 256, 0, stream>>>(l1x[s2], l2x[s2], l1p[s2],
                                                                     ballIdx, bufA, 512, 128, 64,
                                                                     128, totG2);
        run_mlp_layer(bufA, sa2l[0], bufB, stats, totG2, 131, 128, stream);
        run_mlp_layer(bufB, sa2l[1], bufA, stats, totG2, 128, 128, stream);
        run_mlp_layer(bufA, sa2l[2], bufB, stats, totG2, 128, 256, stream);
        int totP2 = 16 * 128 * 256;
        maxpool_kernel<<<(totP2 + 255) / 256, 256, 0, stream>>>(bufB, l2p[s2], 64, 256, totP2);
    }

    // ---------------------- cross attention (both dirs) ---------------------
    for (int s2 = 0; s2 < 2; ++s2) {
        const float* X1 = l2p[s2];
        const float* X2 = l2p[1 - s2];
        dim3 g1((2048 + 127) / 128, (256 + 63) / 64);
        gemm_bias_kernel<<<g1, 256, 0, stream>>>(X1, Wq, nullptr, qb, 2048, 256, 256);
        gemm_bias_kernel<<<g1, 256, 0, stream>>>(X2, Wk, nullptr, kb, 2048, 256, 256);
        gemm_bias_kernel<<<g1, 256, 0, stream>>>(X2, Wv, nullptr, vb, 2048, 256, 256);
        int totSc = 16 * 128 * 128;
        attn_scores_kernel<<<(totSc + 255) / 256, 256, 0, stream>>>(qb, kb, scb, 128, 128, 256,
                                                                    0.0625f, totSc);
        softmax_rows_kernel<<<16 * 128, 128, 0, stream>>>(scb, 128);
        int totAv = 2048 * 256;
        attn_apply_kernel<<<(totAv + 255) / 256, 256, 0, stream>>>(scb, vb, avb, 128, 128, 256, totAv);
        gemm_bias_kernel<<<g1, 256, 0, stream>>>(avb, Wo, nullptr, aob, 2048, 256, 256);
        concat_rows_kernel<<<(2048 + 255) / 256, 256, 0, stream>>>(X1, aob, l2c[s2], 256, 256, 2048);
    }

    // ----------------------- feature propagation ----------------------------
    float* outP[2] = { (float*)d_out, (float*)d_out + (size_t)16 * 4096 * 128 };
    for (int s2 = 0; s2 < 2; ++s2) {
        // fp2: (l1x vs l2x), pts1=l1p(128), pts2=l2c(512) -> 640 -> 512 -> 256
        int tot1 = 16 * 512;
        three_interp_kernel<<<(tot1 + 255) / 256, 256, 0, stream>>>(l1x[s2], l2x[s2], l2c[s2],
                                                                    bufB, 512, 128, 512, tot1);
        concat_rows_kernel<<<(tot1 + 255) / 256, 256, 0, stream>>>(l1p[s2], bufB, bufA, 128, 512, tot1);
        run_mlp_layer(bufA, fp2l[0], bufB, stats, 8192, 640, 512, stream);
        run_mlp_layer(bufB, fp2l[1], l1b,  stats, 8192, 512, 256, stream);

        // fp1: (l0x vs l1x), pts2=l1b(256) -> 256 -> 128 -> 128, final into d_out
        int tot0 = 16 * 4096;
        three_interp_kernel<<<(tot0 + 255) / 256, 256, 0, stream>>>(xyzIn[s2], l1x[s2], l1b,
                                                                    bufA, 4096, 512, 256, tot0);
        run_mlp_layer(bufA, fp1l[0], bufB, stats, 65536, 256, 256, stream);
        run_mlp_layer(bufB, fp1l[1], bufA, stats, 65536, 256, 128, stream);
        run_mlp_layer(bufA, fp1l[2], outP[s2], stats, 65536, 128, 128, stream);
    }
}
